// MolecularGNN_33002528702486
// MI455X (gfx1250) — matrix operand order repaired and checked by reading, not compiled, not run on hardware
//
#include <hip/hip_runtime.h>
#include <hip/hip_bf16.h>
#include <math.h>

// ---------------------------------------------------------------------------
// MolecularGNN (3-layer GATv2 + BN/ELU + pooling + MLP) for gfx1250 (MI455X)
//  - dense transforms: v_wmma_f32_16x16x32_bf16, 32x64 per wave (8 WMMA/K-tile)
//  - LDS staging via TDM tensor_load_to_lds (D# descriptor, 6-arg builtin)
//  - edge scatter-softmax: bandwidth-bound VALU + atomics
// ---------------------------------------------------------------------------

typedef __bf16 bf16_t;
typedef __attribute__((ext_vector_type(8)))  bf16_t bf16x8;
typedef __attribute__((ext_vector_type(16))) bf16_t bf16x16;
typedef __attribute__((ext_vector_type(8)))  float  f32x8;
typedef __attribute__((ext_vector_type(4)))  unsigned int u32x4;
typedef __attribute__((ext_vector_type(8)))  int i32x8;
typedef __attribute__((ext_vector_type(4)))  int i32x4;

#ifndef USE_TDM
#define USE_TDM 1
#endif
#if USE_TDM && defined(__has_builtin)
#if __has_builtin(__builtin_amdgcn_tensor_load_to_lds) && \
    __has_builtin(__builtin_amdgcn_s_wait_tensorcnt)
#define TDM_OK 1
#endif
#endif

#define NN        60000
#define EE        240000
#define EA        (EE + NN)      // edges + self loops
#define GBATCH    3000
#define FNODE     73
#define FNODE_PAD 96             // pad K to multiple of 32
#define FEDGE     10
#define HIDC      128
#define TASKS     12

// ---------------- helpers ----------------
__device__ __forceinline__ bf16_t us_bf(unsigned short u) {
  union { unsigned short u; bf16_t b; } x; x.u = u; return x.b;
}
__device__ __forceinline__ bf16_t f2bf(float f) {   // round-to-nearest-even
  unsigned u = __float_as_uint(f);
  u = (u + 0x7FFFu + ((u >> 16) & 1u)) >> 16;
  return us_bf((unsigned short)u);
}
__device__ __forceinline__ float atomicMaxF(float* addr, float v) {
  if (v >= 0.f) return __int_as_float(atomicMax((int*)addr, __float_as_int(v)));
  return __uint_as_float(atomicMin((unsigned int*)addr, __float_as_uint(v)));
}
__device__ __forceinline__ float act_apply(float x, int act) {
  if (act == 1) return x > 0.f ? x : __expf(x) - 1.f;   // ELU
  if (act == 2) return fmaxf(x, 0.f);                   // ReLU
  return x;
}
__device__ __forceinline__ bf16x16 frag16(const bf16_t* p0, const bf16_t* p1) {
  bf16x8 lo = *(const bf16x8*)p0;
  bf16x8 hi = *(const bf16x8*)p1;
  return __builtin_shufflevector(lo, hi, 0,1,2,3,4,5,6,7,8,9,10,11,12,13,14,15);
}

#ifdef TDM_OK
// Issue one 2-D TDM tile load: global[rows x 32cols of bf16] -> LDS (packed).
// tensor_dim* are clipped so out-of-range rows/cols read as zero (ISA 8.2).
// 6-arg builtin (amdgpu-toolchain clang-23):
//   (u32x4 g0, i32x8 g1, i32x4 g2, i32x4 g3, i32x8 extra, i32 cpol)
__device__ __forceinline__ void tdm_load_2d(unsigned lds_off, const bf16_t* gsrc,
                                            unsigned tile_rows, unsigned tile_cols,
                                            unsigned rows_left, unsigned cols_left,
                                            unsigned row_stride_elems) {
  unsigned long long ga = (unsigned long long)(size_t)gsrc;
  u32x4 g0;
  g0[0] = 1u;                                   // count=1, user mode
  g0[1] = lds_off;                              // lds_addr (bytes)
  g0[2] = (unsigned)(ga & 0xFFFFFFFFu);         // global_addr[31:0]
  g0[3] = (unsigned)((ga >> 32) & 0x01FFFFFFu)  // global_addr[56:32]
          | 0x80000000u;                        // type=2 ("image")
  i32x8 g1;
  g1[0] = (int)(1u << 16);                      // data_size=1 (2B), mask=0
  g1[1] = (int)((cols_left & 0xFFFFu) << 16);   // tensor_dim0[15:0]
  g1[2] = (int)(((cols_left >> 16) & 0xFFFFu) | ((rows_left & 0xFFFFu) << 16));
  g1[3] = (int)(((rows_left >> 16) & 0xFFFFu) | ((tile_cols & 0xFFFFu) << 16));
  g1[4] = (int)(tile_rows & 0xFFFFu);           // tile_dim1; tile_dim2=0
  g1[5] = (int)row_stride_elems;                // tensor_dim0_stride[31:0]
  g1[6] = 0;                                    // stride hi / dim1_stride lo
  g1[7] = 0;
  i32x4 z4; z4[0] = 0; z4[1] = 0; z4[2] = 0; z4[3] = 0;
  i32x8 z8;
#pragma unroll
  for (int j = 0; j < 8; ++j) z8[j] = 0;
  __builtin_amdgcn_tensor_load_to_lds(g0, g1, z4, z4, z8, 0);
}
#endif

// ---------------- f32 -> bf16 convert (activations, zero-padded) ----------
__global__ void k_cvt_pad(const float* __restrict__ src, bf16_t* __restrict__ dst,
                          int sr, int sc, int dr, int dc) {
  int i = blockIdx.x * blockDim.x + threadIdx.x;
  if (i >= dr * dc) return;
  int r = i / dc, c = i - r * dc;
  float v = (r < sr && c < sc) ? src[(size_t)r * sc + c] : 0.f;
  dst[i] = f2bf(v);
}

// weights: f32 [K][N] -> bf16 transposed [Np][Kp] (zero padded)
__global__ void k_cvt_wT(const float* __restrict__ src, bf16_t* __restrict__ dst,
                         int K, int N, int Kp, int Np) {
  int i = blockIdx.x * blockDim.x + threadIdx.x;
  if (i >= Np * Kp) return;
  int n = i / Kp, k = i - n * Kp;
  float v = (n < N && k < K) ? src[(size_t)k * N + n] : 0.f;
  dst[i] = f2bf(v);
}

__global__ void k_fill_f32(float* __restrict__ p, float v, int n) {
  int i = blockIdx.x * blockDim.x + threadIdx.x;
  if (i < n) p[i] = v;
}

// ---------------- WMMA GEMM ----------------
// C[MxN] = act(A[MxK]bf16 @ Bt^T + bias), Bt is [Np][K] (pre-transposed, padded)
// Block: 256 threads (8 waves), tile 256x64, wave -> 32x64 (8 WMMA / K-tile).
#define GMT 256
#define GNT 64
#define GKT 32

__global__ __launch_bounds__(256)
void k_gemm_bf16(const bf16_t* __restrict__ A, const bf16_t* __restrict__ Bt,
                 const float* __restrict__ bias, float* __restrict__ C,
                 bf16_t* __restrict__ Cbf, int M, int N, int K, int act) {
  __shared__ __align__(16) bf16_t As[GMT][GKT];   // 16 KB
  __shared__ __align__(16) bf16_t Bs[GNT][GKT];   //  4 KB  ([n][k])
  const int tid  = threadIdx.x;
  const int wave = tid >> 5;
  const int lane = tid & 31;
  const int m0 = blockIdx.y * GMT;
  const int n0 = blockIdx.x * GNT;

  f32x8 acc[2][4];
#pragma unroll
  for (int h = 0; h < 2; ++h)
#pragma unroll
    for (int i = 0; i < 4; ++i)
#pragma unroll
      for (int e = 0; e < 8; ++e) acc[h][i][e] = 0.f;

  for (int k0 = 0; k0 < K; k0 += GKT) {
    __syncthreads();   // previous compute done before LDS overwrite
#ifdef TDM_OK
    if (wave == 0) {   // one wave drives the Tensor Data Mover for both tiles
      unsigned rows_left = (unsigned)(M - m0);
      tdm_load_2d((unsigned)(size_t)&As[0][0], A + (size_t)m0 * K + k0,
                  GMT, GKT, rows_left, (unsigned)(K - k0), (unsigned)K);
      tdm_load_2d((unsigned)(size_t)&Bs[0][0], Bt + (size_t)n0 * K + k0,
                  GNT, GKT, (unsigned)GNT, (unsigned)(K - k0), (unsigned)K);
      __builtin_amdgcn_s_wait_tensorcnt((short)0);
    }
#else
    {   // A tile: one row of 32 bf16 per thread (4x b128)
      int gm = m0 + tid;
      bf16_t* d = &As[tid][0];
      if (gm < M) {
        const bf16_t* s = A + (size_t)gm * K + k0;
#pragma unroll
        for (int j = 0; j < 4; ++j)
          *(bf16x8*)(d + 8 * j) = *(const bf16x8*)(s + 8 * j);
      } else {
#pragma unroll
        for (int j = 0; j < 32; ++j) d[j] = us_bf(0);
      }
    }
    if (tid < 128) {   // B tile: [n][k] rows are contiguous in Bt (2x b128)
      int n = tid >> 1, half = (tid & 1) * 16;
      const bf16_t* s = Bt + (size_t)(n0 + n) * K + k0 + half;
      bf16_t* d = &Bs[n][half];
      *(bf16x8*)d       = *(const bf16x8*)s;
      *(bf16x8*)(d + 8) = *(const bf16x8*)(s + 8);
    }
#endif
    __syncthreads();

    // A frag per ISA layout: lane<16 -> K0-7,K16-23 of row(lane); lane>=16 -> +8
    const int akb = (lane >> 4) * 8;
    const int bkb = (lane >> 4) * 8;     // B: lane<16 -> K0-7,K16-23; lane>=16 -> +8
    const bf16_t* ar0 = &As[wave * 32 + (lane & 15)][0];
    const bf16_t* ar1 = &As[wave * 32 + 16 + (lane & 15)][0];
    bf16x16 a0 = frag16(ar0 + akb, ar0 + akb + 16);
    bf16x16 a1 = frag16(ar1 + akb, ar1 + akb + 16);
#pragma unroll
    for (int nf = 0; nf < 4; ++nf) {
      const bf16_t* bc = &Bs[nf * 16 + (lane & 15)][bkb];
      bf16x16 bfrag = frag16(bc, bc + 16);
      acc[0][nf] = __builtin_amdgcn_wmma_f32_16x16x32_bf16(
          false, a0, false, bfrag, (short)0, acc[0][nf], false, false);
      acc[1][nf] = __builtin_amdgcn_wmma_f32_16x16x32_bf16(
          false, a1, false, bfrag, (short)0, acc[1][nf], false, false);
    }
  }

  // epilogue: D layout lane<16 -> M=v, lane>=16 -> M=v+8; N=lane&15
  int ncol = n0 + (lane & 15);
#pragma unroll
  for (int h = 0; h < 2; ++h) {
    int mrow = m0 + wave * 32 + h * 16 + ((lane >= 16) ? 8 : 0);
#pragma unroll
    for (int nf = 0; nf < 4; ++nf) {
      int n = ncol + nf * 16;
      if (n >= N) continue;
      float bv = bias ? bias[n] : 0.f;
#pragma unroll
      for (int v = 0; v < 8; ++v) {
        int m = mrow + v;
        if (m < M) {
          float x = act_apply(acc[h][nf][v] + bv, act);
          C[(size_t)m * N + n] = x;
          if (Cbf) Cbf[(size_t)m * N + n] = f2bf(x);
        }
      }
    }
  }
}

// ---------------- edge phase ----------------
__global__ void k_deg_loopea(const int* __restrict__ ei, const float* __restrict__ ea,
                             float* __restrict__ deg, float* __restrict__ lea) {
  int e = blockIdx.x * blockDim.x + threadIdx.x;
  if (e >= EE) return;
  int d = ei[EE + e];
  atomicAdd(&deg[d], 1.f);
#pragma unroll
  for (int f = 0; f < FEDGE; ++f)
    atomicAdd(&lea[(size_t)d * FEDGE + f], ea[(size_t)e * FEDGE + f]);
}
__global__ void k_loopea_div(float* __restrict__ lea, const float* __restrict__ deg) {
  int i = blockIdx.x * blockDim.x + threadIdx.x;
  if (i >= NN * FEDGE) return;
  lea[i] /= fmaxf(deg[i / FEDGE], 1.f);
}

// attention logits: one wave per (edge, head); We staged in LDS
__global__ __launch_bounds__(256)
void k_edge_alpha(const int* __restrict__ ei, const float* __restrict__ eattr,
                  const float* __restrict__ lea,
                  const float* __restrict__ xl, const float* __restrict__ xr,
                  const float* __restrict__ We, const float* __restrict__ att,
                  float* __restrict__ alpha, float* __restrict__ amax, int H) {
  const int HC = H * HIDC;
  __shared__ float WeS[FEDGE * 512];
  for (int i = threadIdx.x; i < FEDGE * HC; i += blockDim.x) WeS[i] = We[i];
  __syncthreads();

  int wid  = blockIdx.x * 8 + (threadIdx.x >> 5);
  int lane = threadIdx.x & 31;
  if (wid >= EA * H) return;
  int e = wid / H, h = wid - e * H;

  int s, d; const float* ea;
  if (e < EE) { s = ei[e]; d = ei[EE + e]; ea = eattr + (size_t)e * FEDGE; }
  else        { s = d = e - EE;            ea = lea   + (size_t)(e - EE) * FEDGE; }

  float eav[FEDGE];
#pragma unroll
  for (int f = 0; f < FEDGE; ++f) eav[f] = ea[f];

  const float* xls  = xl  + (size_t)s * HC + h * HIDC;
  const float* xrd  = xr  + (size_t)d * HC + h * HIDC;
  const float* atth = att + h * HIDC;
  float part = 0.f;
#pragma unroll
  for (int j = 0; j < 4; ++j) {
    int c = lane + j * 32;
    float ee = 0.f;
#pragma unroll
    for (int f = 0; f < FEDGE; ++f) ee += eav[f] * WeS[f * HC + h * HIDC + c];
    float m = xls[c] + xrd[c] + ee;
    m = (m > 0.f) ? m : 0.2f * m;                 // leaky_relu(0.2)
    part += atth[c] * m;
  }
#pragma unroll
  for (int off = 16; off > 0; off >>= 1) part += __shfl_xor(part, off);
  if (lane == 0) {
    alpha[wid] = part;
    atomicMaxF(&amax[d * H + h], part);
  }
}

__global__ void k_edge_expsum(const int* __restrict__ ei, float* __restrict__ alpha,
                              const float* __restrict__ amax, float* __restrict__ asum,
                              int H) {
  int i = blockIdx.x * blockDim.x + threadIdx.x;
  if (i >= EA * H) return;
  int e = i / H, h = i - e * H;
  int d = (e < EE) ? ei[EE + e] : (e - EE);
  float ex = __expf(alpha[i] - amax[d * H + h]);
  alpha[i] = ex;
  atomicAdd(&asum[d * H + h], ex);
}

__global__ __launch_bounds__(256)
void k_edge_scatter(const int* __restrict__ ei, const float* __restrict__ alpha,
                    const float* __restrict__ asum, const float* __restrict__ xl,
                    float* __restrict__ out, int H) {
  int e    = blockIdx.x * 8 + (threadIdx.x >> 5);
  int lane = threadIdx.x & 31;
  if (e >= EA) return;
  const int HC = H * HIDC;
  int s, d;
  if (e < EE) { s = ei[e]; d = ei[EE + e]; } else { s = d = e - EE; }
  for (int h = 0; h < H; ++h) {
    float a = alpha[e * H + h] / asum[d * H + h];
    const float* xs = xl  + (size_t)s * HC + h * HIDC;
    float*       od = out + (size_t)d * HC + h * HIDC;
#pragma unroll
    for (int j = 0; j < 4; ++j) {
      int c = lane + j * 32;
      atomicAdd(&od[c], xs[c] * a);
    }
  }
}

// ---------------- batch-norm (GAT out-bias cancels inside BN) ----------
#define CS_ROWS 1024
__global__ void k_colstats(const float* __restrict__ X, float* __restrict__ musum,
                           float* __restrict__ sqsum, int outd) {
  int r0 = blockIdx.x * CS_ROWS;
  int r1 = min(r0 + CS_ROWS, NN);
  for (int c = threadIdx.x; c < outd; c += blockDim.x) {
    float s = 0.f, q = 0.f;
    for (int r = r0; r < r1; ++r) {
      float v = X[(size_t)r * outd + c];
      s += v; q += v * v;
    }
    atomicAdd(&musum[c], s);
    atomicAdd(&sqsum[c], q);
  }
}
__global__ void k_stats_final(const float* __restrict__ musum, const float* __restrict__ sqsum,
                              float* __restrict__ mu, float* __restrict__ rstd, int outd) {
  int c = threadIdx.x + blockIdx.x * blockDim.x;
  if (c >= outd) return;
  float m   = musum[c] / (float)NN;
  float var = sqsum[c] / (float)NN - m * m;
  mu[c]   = m;
  rstd[c] = rsqrtf(fmaxf(var, 0.f) + 1e-5f);
}
__global__ void k_bn_elu(float* __restrict__ X, bf16_t* __restrict__ Xbf,
                         const float* __restrict__ mu, const float* __restrict__ rstd,
                         const float* __restrict__ gamma, const float* __restrict__ beta,
                         int outd) {
  size_t i = (size_t)blockIdx.x * blockDim.x + threadIdx.x;
  if (i >= (size_t)NN * outd) return;
  int c = (int)(i % outd);
  float v = X[i];
  float x = gamma[c] * (v - mu[c]) * rstd[c] + beta[c];
  x = x > 0.f ? x : __expf(x) - 1.f;
  X[i]   = x;
  Xbf[i] = f2bf(x);
}

// ---------------- graph pooling ----------------
__global__ void k_batch_accum(const float* __restrict__ h, const int* __restrict__ batch,
                              float* __restrict__ gsum, float* __restrict__ gmax,
                              float* __restrict__ cnt) {
  size_t i = (size_t)blockIdx.x * blockDim.x + threadIdx.x;
  if (i >= (size_t)NN * HIDC) return;
  int n = (int)(i >> 7), c = (int)(i & 127);
  int b = batch[n];
  float v = h[i];
  atomicAdd(&gsum[(size_t)b * HIDC + c], v);
  atomicMaxF(&gmax[(size_t)b * HIDC + c], v);
  if (c == 0) atomicAdd(&cnt[b], 1.f);
}
__global__ void k_batch_final(const float* __restrict__ gsum, const float* __restrict__ gmax,
                              const float* __restrict__ cnt, float* __restrict__ g,
                              bf16_t* __restrict__ gbf) {
  int i = blockIdx.x * blockDim.x + threadIdx.x;
  if (i >= GBATCH * HIDC) return;
  int b = i >> 7, c = i & 127;
  float mean = gsum[i] / fmaxf(cnt[b], 1.f);
  float mx   = gmax[i];
  if (mx < -1e30f) mx = 0.f;           // isfinite(gmax) ? gmax : 0
  g[(size_t)b * 256 + c]         = mean;
  g[(size_t)b * 256 + 128 + c]   = mx;
  gbf[(size_t)b * 256 + c]       = f2bf(mean);
  gbf[(size_t)b * 256 + 128 + c] = f2bf(mx);
}

// ---------------------------------------------------------------------------
extern "C" void kernel_launch(void* const* d_in, const int* in_sizes, int n_in,
                              void* d_out, int out_size, void* d_ws, size_t ws_size,
                              hipStream_t stream) {
  (void)in_sizes; (void)n_in; (void)out_size; (void)ws_size;

  const float* x     = (const float*)d_in[0];
  const int*   ei    = (const int*)  d_in[1];
  const float* eattr = (const float*)d_in[2];
  const int*   batch = (const int*)  d_in[3];
  const float* W_in  = (const float*)d_in[4];
  const float* b_in  = (const float*)d_in[5];
  const float* W1 = (const float*)d_in[33];
  const float* b1 = (const float*)d_in[34];
  const float* W2 = (const float*)d_in[35];
  const float* b2 = (const float*)d_in[36];
  const float* W3 = (const float*)d_in[37];
  const float* b3 = (const float*)d_in[38];
  float* out = (float*)d_out;

  // ---- carve workspace (~460 MB) ----
  char* p = (char*)d_ws;
  auto carve = [&](size_t bytes) -> char* {
    char* r = p; p += (bytes + 255) & ~(size_t)255; return r;
  };
  float*  h     = (float*) carve((size_t)NN * 512 * 4);   // layer io / scatter out
  bf16_t* hbf   = (bf16_t*)carve((size_t)NN * 512 * 2);
  float*  xl    = (float*) carve((size_t)NN * 512 * 4);
  float*  xr    = (float*) carve((size_t)NN * 512 * 4);
  bf16_t* xpad  = (bf16_t*)carve((size_t)NN * FNODE_PAD * 2);
  float*  alpha = (float*) carve((size_t)EA * 4 * 4);
  float*  amax  = (float*) carve((size_t)NN * 4 * 4);
  float*  asum  = (float*) carve((size_t)NN * 4 * 4);
  float*  deg   = (float*) carve((size_t)NN * 4);
  float*  lea   = (float*) carve((size_t)NN * FEDGE * 4);
  float*  musum = (float*) carve(512 * 4);
  float*  sqsum = (float*) carve(512 * 4);
  float*  mu    = (float*) carve(512 * 4);
  float*  rstd  = (float*) carve(512 * 4);
  bf16_t* wbuf  = (bf16_t*)carve((size_t)512 * 512 * 2);  // transposed bf16 weights
  float*  cnt   = (float*) carve((size_t)GBATCH * 4);
  float*  gsum  = (float*) carve((size_t)GBATCH * HIDC * 4);
  float*  gmax  = (float*) carve((size_t)GBATCH * HIDC * 4);
  float*  g     = (float*) carve((size_t)GBATCH * 256 * 4);
  bf16_t* gbf   = (bf16_t*)carve((size_t)GBATCH * 256 * 2);
  float*  t1    = (float*) carve((size_t)GBATCH * 256 * 4);
  bf16_t* t1bf  = (bf16_t*)carve((size_t)GBATCH * 256 * 2);
  float*  t2    = (float*) carve((size_t)GBATCH * 128 * 4);
  bf16_t* t2bf  = (bf16_t*)carve((size_t)GBATCH * 128 * 2);

  auto blk = [](size_t n, int t) { return (unsigned)((n + t - 1) / t); };
  auto gemm = [&](const bf16_t* A, const bf16_t* Bt, const float* bias,
                  float* C, bf16_t* Cbf, int M, int N, int K, int act) {
    dim3 grid((N + GNT - 1) / GNT, (M + GMT - 1) / GMT);
    k_gemm_bf16<<<grid, 256, 0, stream>>>(A, Bt, bias, C, Cbf, M, N, K, act);
  };
  // weight W[K][N] f32 -> wbuf[Np][Kp] bf16 transposed, Np = ceil(N/64)*64
  auto cvtw = [&](const float* W, int K, int N, int Kp) {
    int Np = ((N + GNT - 1) / GNT) * GNT;
    k_cvt_wT<<<blk((size_t)Np * Kp, 256), 256, 0, stream>>>(W, wbuf, K, N, Kp, Np);
  };

  // ---- self-loop edge_attr = per-target mean (reused by all layers) ----
  (void)hipMemsetAsync(deg, 0, (size_t)NN * 4, stream);
  (void)hipMemsetAsync(lea, 0, (size_t)NN * FEDGE * 4, stream);
  k_deg_loopea<<<blk(EE, 256), 256, 0, stream>>>(ei, eattr, deg, lea);
  k_loopea_div<<<blk((size_t)NN * FEDGE, 256), 256, 0, stream>>>(lea, deg);

  // ---- input projection: h = elu(x @ W_in + b_in)  (K padded 73 -> 96) ----
  k_cvt_pad<<<blk((size_t)NN * FNODE_PAD, 256), 256, 0, stream>>>(
      x, xpad, NN, FNODE, NN, FNODE_PAD);
  cvtw(W_in, FNODE, HIDC, FNODE_PAD);
  gemm(xpad, wbuf, b_in, h, hbf, NN, HIDC, FNODE_PAD, /*ELU*/1);

  // ---- GATv2 layers ----
  const int ind_l[3] = {128, 512, 512};
  const int H_l[3]   = {4, 4, 1};
  for (int i = 0; i < 3; ++i) {
    const int base = 6 + i * 9;
    const float* Wl  = (const float*)d_in[base + 0];
    const float* bl  = (const float*)d_in[base + 1];
    const float* Wr  = (const float*)d_in[base + 2];
    const float* br  = (const float*)d_in[base + 3];
    const float* We  = (const float*)d_in[base + 4];
    const float* att = (const float*)d_in[base + 5];
    const float* gam = (const float*)d_in[base + 7];
    const float* bet = (const float*)d_in[base + 8];
    const int ind = ind_l[i], H = H_l[i], HC = H * HIDC;  // outd == HC

    cvtw(Wl, ind, HC, ind);
    gemm(hbf, wbuf, bl, xl, nullptr, NN, HC, ind, 0);
    cvtw(Wr, ind, HC, ind);
    gemm(hbf, wbuf, br, xr, nullptr, NN, HC, ind, 0);

    // edge phase init (h is dead now; reuse as scatter accumulator)
    (void)hipMemsetAsync(h,    0, (size_t)NN * HC * 4, stream);
    (void)hipMemsetAsync(asum, 0, (size_t)NN * H * 4, stream);
    k_fill_f32<<<blk((size_t)NN * H, 256), 256, 0, stream>>>(
        amax, -__builtin_inff(), NN * H);

    k_edge_alpha<<<blk((size_t)EA * H * 32, 256), 256, 0, stream>>>(
        ei, eattr, lea, xl, xr, We, att, alpha, amax, H);
    k_edge_expsum<<<blk((size_t)EA * H, 256), 256, 0, stream>>>(
        ei, alpha, amax, asum, H);
    k_edge_scatter<<<blk((size_t)EA * 32, 256), 256, 0, stream>>>(
        ei, alpha, asum, xl, h, H);

    // BatchNorm (+ELU)
    (void)hipMemsetAsync(musum, 0, 512 * 4, stream);
    (void)hipMemsetAsync(sqsum, 0, 512 * 4, stream);
    k_colstats<<<blk(NN, CS_ROWS), 256, 0, stream>>>(h, musum, sqsum, HC);
    k_stats_final<<<2, 256, 0, stream>>>(musum, sqsum, mu, rstd, HC);
    k_bn_elu<<<blk((size_t)NN * HC, 256), 256, 0, stream>>>(
        h, hbf, mu, rstd, gam, bet, HC);
  }

  // ---- graph pooling: g = [mean, max] over batch ----
  (void)hipMemsetAsync(gsum, 0, (size_t)GBATCH * HIDC * 4, stream);
  (void)hipMemsetAsync(cnt,  0, (size_t)GBATCH * 4, stream);
  k_fill_f32<<<blk((size_t)GBATCH * HIDC, 256), 256, 0, stream>>>(
      gmax, -__builtin_inff(), GBATCH * HIDC);
  k_batch_accum<<<blk((size_t)NN * HIDC, 256), 256, 0, stream>>>(
      h, batch, gsum, gmax, cnt);
  k_batch_final<<<blk((size_t)GBATCH * HIDC, 256), 256, 0, stream>>>(
      gsum, gmax, cnt, g, gbf);

  // ---- MLP head (WMMA GEMMs) ----
  cvtw(W1, 256, 256, 256);
  gemm(gbf,  wbuf, b1, t1, t1bf, GBATCH, 256, 256, /*ReLU*/2);
  cvtw(W2, 256, 128, 256);
  gemm(t1bf, wbuf, b2, t2, t2bf, GBATCH, 128, 256, /*ReLU*/2);
  cvtw(W3, 128, TASKS, 128);
  gemm(t2bf, wbuf, b3, out, nullptr, GBATCH, TASKS, 128, 0);
}
